// WALinear_online_68324339745161
// MI455X (gfx1250) — 8-bit matrix operand order repaired and checked by reading, compiled, not run on hardware
//
#include <hip/hip_runtime.h>
#include <hip/hip_bf16.h>
#include <stdint.h>


#define K_DIM 4096
#define N_DIM 11008
#define M_TOT 8192
#define GRP   128
#define NGRP  (K_DIM / GRP)

typedef int   v8i __attribute__((ext_vector_type(8)));
typedef float v8f __attribute__((ext_vector_type(8)));
typedef int   v2i __attribute__((ext_vector_type(2)));

__global__ __launch_bounds__(256)
void wa_quant_act(const float* __restrict__ x,
                  int8_t* __restrict__ xq,
                  float* __restrict__ ascale)
{
    const int token = blockIdx.x;
    const int tid   = threadIdx.x;
    const float* row = x + (size_t)token * K_DIM;

    __shared__ float red[256];
    __shared__ float s_inv;

    float4 vals[4];
    float amax = 0.0f;
#pragma unroll
    for (int u = 0; u < 4; ++u) {
        const float4 v = *(const float4*)(row + (size_t)(u * 256 + tid) * 4);
        vals[u] = v;
        amax = fmaxf(amax,
               fmaxf(fmaxf(fabsf(v.x), fabsf(v.y)),
                     fmaxf(fabsf(v.z), fabsf(v.w))));
    }
    red[tid] = amax;
    __syncthreads();
#pragma unroll
    for (int s = 128; s > 0; s >>= 1) {
        if (tid < s) red[tid] = fmaxf(red[tid], red[tid + s]);
        __syncthreads();
    }
    if (tid == 0) {
        const float a  = fmaxf(red[0], 1e-5f);
        const float sc = a * (1.0f / 127.0f);
        ascale[token]  = sc;
        s_inv          = 1.0f / sc;
    }
    __syncthreads();

    const float inv = s_inv;
    int* qout = (int*)(xq + (size_t)token * K_DIM);
#pragma unroll
    for (int u = 0; u < 4; ++u) {
        int q0 = (int)rintf(vals[u].x * inv);
        int q1 = (int)rintf(vals[u].y * inv);
        int q2 = (int)rintf(vals[u].z * inv);
        int q3 = (int)rintf(vals[u].w * inv);
        q0 = min(127, max(-127, q0));
        q1 = min(127, max(-127, q1));
        q2 = min(127, max(-127, q2));
        q3 = min(127, max(-127, q3));
        qout[u * 256 + tid] = (q0 & 0xFF) | ((q1 & 0xFF) << 8) |
                              ((q2 & 0xFF) << 16) | (q3 << 24);
    }
}

__global__ __launch_bounds__(256)
void wa_w4a8_wmma_gemm(const int8_t* __restrict__ xq,
                       const float*  __restrict__ ascale,
                       const int*    __restrict__ qweight,
                       const float*  __restrict__ wscale,
                       const int*    __restrict__ wzero,
                       const float*  __restrict__ bias,
                       float*        __restrict__ out)
{
    __shared__ __align__(16) int8_t lds_b[64 * 64];

    const int tid    = threadIdx.x;
    const int lane   = tid & 31;
    const int wave   = tid >> 5;
    const int wave_m = wave & 3;
    const int wave_n = wave >> 2;

    const int n_tile = blockIdx.x;
    const int m_tile = blockIdx.y;

    const int m_base = m_tile * 128 + wave_m * 32;
    const int n_base = n_tile * 64  + wave_n * 32;

    const int s_nloc = tid >> 2;
    const int s_o    = n_tile * 64 + s_nloc;
    const int s_quad = tid & 3;
    const int s_nl   = s_nloc & 15;
    int8_t* s_dst0   = lds_b + (s_nloc >> 4) * 1024;

    const int nl   = lane & 15;
    const int half = lane >> 4;

    v8f facc[2][2];
#pragma unroll
    for (int a = 0; a < 2; ++a)
#pragma unroll
        for (int b = 0; b < 2; ++b)
            facc[a][b] = (v8f){0.f,0.f,0.f,0.f,0.f,0.f,0.f,0.f};

    for (int g = 0; g < NGRP; ++g) {
        const int zp = wzero[s_o * NGRP + g];

        v8i iacc[2][2];
#pragma unroll
        for (int a = 0; a < 2; ++a)
#pragma unroll
            for (int b = 0; b < 2; ++b)
                iacc[a][b] = (v8i){0,0,0,0,0,0,0,0};

#pragma unroll
        for (int kk = 0; kk < 2; ++kk) {
            const int k0 = g * GRP + kk * 64;

            __syncthreads();
            {
                const int* qrow = qweight + (size_t)s_o * (K_DIM / 2) + (k0 >> 1);
#pragma unroll
                for (int u = 0; u < 8; ++u) {
                    const int joff = s_quad * 8 + u;
                    const int qv   = qrow[joff];
                    const int kl   = joff << 1;
                    const int bhalf = (kl >> 4) & 1;
                    const int r     = ((kl >> 5) << 2) | ((kl >> 2) & 3);
                    const int lane_t = bhalf * 16 + s_nl;
                    const int off    = lane_t * 32 + r * 4 + (kl & 3);
                    const int lo = (qv & 15) - zp;
                    const int hi = ((qv >> 4) & 15) - zp;
                    const unsigned short pk =
                        (unsigned short)((lo & 0xFF) | ((hi & 0xFF) << 8));
                    *(unsigned short*)(s_dst0 + off) = pk;
                }
            }
            __syncthreads();

            v8i bf[2];
#pragma unroll
            for (int ns = 0; ns < 2; ++ns)
                bf[ns] = *(const v8i*)(lds_b + (wave_n * 2 + ns) * 1024 + lane * 32);

            v8i af[2];
#pragma unroll
            for (int ms = 0; ms < 2; ++ms) {
                const int8_t* row = xq +
                    (size_t)(m_base + ms * 16 + nl) * K_DIM + k0 + half * 16;
                const v2i q0 = *(const v2i*)(row +  0);
                const v2i q1 = *(const v2i*)(row +  8);
                const v2i q2 = *(const v2i*)(row + 32);
                const v2i q3 = *(const v2i*)(row + 40);
                af[ms] = (v8i){q0.x, q0.y, q1.x, q1.y, q2.x, q2.y, q3.x, q3.y};
            }

#pragma unroll
            for (int ms = 0; ms < 2; ++ms)
#pragma unroll
                for (int ns = 0; ns < 2; ++ns)
                    iacc[ms][ns] = __builtin_amdgcn_wmma_i32_16x16x64_iu8(
                        true, af[ms], true, bf[ns], iacc[ms][ns], false, false);
        }

#pragma unroll
        for (int ns = 0; ns < 2; ++ns) {
            const int   n = n_base + ns * 16 + nl;
            const float s = wscale[n * NGRP + g];
#pragma unroll
            for (int ms = 0; ms < 2; ++ms)
#pragma unroll
                for (int e = 0; e < 8; ++e)
                    facc[ms][ns][e] += s * (float)iacc[ms][ns][e];
        }
    }

#pragma unroll
    for (int ms = 0; ms < 2; ++ms) {
        float as[8];
#pragma unroll
        for (int e = 0; e < 8; ++e)
            as[e] = ascale[m_base + ms * 16 + half * 8 + e];
#pragma unroll
        for (int ns = 0; ns < 2; ++ns) {
            const int   n  = n_base + ns * 16 + nl;
            const float bv = bias[n];
#pragma unroll
            for (int e = 0; e < 8; ++e) {
                const int tok = m_base + ms * 16 + half * 8 + e;
                out[(size_t)tok * N_DIM + n] = as[e] * facc[ms][ns][e] + bv;
            }
        }
    }
}

extern "C" void kernel_launch(void* const* d_in, const int* in_sizes, int n_in,
                              void* d_out, int out_size, void* d_ws, size_t ws_size,
                              hipStream_t stream)
{
    (void)in_sizes; (void)n_in; (void)out_size; (void)ws_size;

    const float* x       = (const float*)d_in[0];
    const int*   qweight = (const int*)  d_in[1];
    const float* wscale  = (const float*)d_in[2];
    const int*   wzero   = (const int*)  d_in[3];
    const float* bias    = (const float*)d_in[4];
    float*       out     = (float*)d_out;

    int8_t* xq     = (int8_t*)d_ws;
    float*  ascale = (float*)((char*)d_ws + (size_t)M_TOT * K_DIM);

    wa_quant_act<<<M_TOT, 256, 0, stream>>>(x, xq, ascale);

    dim3 grid(N_DIM / 64, M_TOT / 128);
    wa_w4a8_wmma_gemm<<<grid, 256, 0, stream>>>(xq, ascale, qweight, wscale,
                                                wzero, bias, out);
}
